// IndexSelect_70858370449709
// MI455X (gfx1250) — matrix operand order repaired and checked by reading, not compiled, not run on hardware
//
#include <hip/hip_runtime.h>
#include <hip/hip_bf16.h>

typedef __attribute__((ext_vector_type(16))) _Float16 v16h;
typedef __attribute__((ext_vector_type(8)))  _Float16 v8h;
typedef __attribute__((ext_vector_type(8)))  float    v8f;

#define NN 50000
#define DD 128
#define EE 800000
#define KSEL 25000
#define SORT_LEN 65536   // 2^16 >= NN
#define NEG_BIG (-3.402823466e38f)

// ---------------------------------------------------------------------------
// WMMA operand loaders (v_wmma_f32_16x16x32_f16, wave32)
// A (16x32 f16): lane l -> row M = l&15; halves: l<16 K={k0+0..7, k0+16..23},
//                l>=16 K={k0+8..15, k0+24..31}  (ISA 7.12.2 16-bit A table)
// B (32x16 f16): lane l -> col N = l&15; same K split as A (l<16 K={k0+0..7, k0+16..23}, l>=16 K={k0+8..15, k0+24..31})
// C/D (16x16 f32): vgpr r, lane l -> M = (l>>4)*8 + r, N = l&15
// ---------------------------------------------------------------------------
__device__ __forceinline__ v16h load_a_f32(const float* base, int row0, int k0, float scale) {
  const int l = threadIdx.x & 31;
  const float* p = base + (size_t)(row0 + (l & 15)) * DD + k0 + ((l >> 4) << 3);
  v16h out;
#pragma unroll
  for (int g = 0; g < 2; ++g) {
    float4 f0 = *(const float4*)(p + g * 16);
    float4 f1 = *(const float4*)(p + g * 16 + 4);
    out[g * 8 + 0] = (_Float16)(f0.x * scale);
    out[g * 8 + 1] = (_Float16)(f0.y * scale);
    out[g * 8 + 2] = (_Float16)(f0.z * scale);
    out[g * 8 + 3] = (_Float16)(f0.w * scale);
    out[g * 8 + 4] = (_Float16)(f1.x * scale);
    out[g * 8 + 5] = (_Float16)(f1.y * scale);
    out[g * 8 + 6] = (_Float16)(f1.z * scale);
    out[g * 8 + 7] = (_Float16)(f1.w * scale);
  }
  return out;
}

// B[k][n] taken from row-major Wrows where Wrows[n][k] supplies column n.
__device__ __forceinline__ v16h load_b_h(const _Float16* Wrows, int n0, int k0) {
  const int l = threadIdx.x & 31;
  const _Float16* p = Wrows + (size_t)(n0 + (l & 15)) * DD + k0 + ((l >> 4) << 3);
  v16h out;
  *(v8h*)&out       = *(const v8h*)p;
  *((v8h*)&out + 1) = *(const v8h*)(p + 16);
  return out;
}

__device__ __forceinline__ float sigm(float x) { return 1.0f / (1.0f + __expf(-x)); }

// ---------------------------------------------------------------------------
// Prep: zero agg/degrees, init sort pad, convert weights to f16 (W_bil transposed)
// ---------------------------------------------------------------------------
__global__ void prep_kernel(const float* Wd, const float* Wg, const float* Wb,
                            _Float16* Wdh, _Float16* Wgh, _Float16* Wbth,
                            float* agg, float* deg_out, float* deg_in,
                            float* keys, int* vals) {
  int i = blockIdx.x * blockDim.x + threadIdx.x;
  if (i < NN * DD) agg[i] = 0.0f;
  if (i < DD * DD) {
    Wdh[i] = (_Float16)Wd[i];
    Wgh[i] = (_Float16)Wg[i];
    int r = i >> 7, c = i & 127;
    Wbth[(size_t)c * DD + r] = (_Float16)Wb[i];  // transpose: column-major access -> rows
  }
  if (i < SORT_LEN) { keys[i] = NEG_BIG; vals[i] = i; }
  if (i < NN) { deg_out[i] = 0.0f; deg_in[i] = 0.0f; }
}

// ---------------------------------------------------------------------------
// Degree accumulation
// ---------------------------------------------------------------------------
__global__ void degree_kernel(const int* src, const int* dst, float* deg_out, float* deg_in) {
  int i = blockIdx.x * blockDim.x + threadIdx.x;
  if (i < EE) {
    atomicAdd(&deg_out[src[i]], 1.0f);
    atomicAdd(&deg_in[dst[i]], 1.0f);
  }
}

// ---------------------------------------------------------------------------
// Dense layer: Y = prelu(X @ W^T + b).  One wave per 16x128 strip, 32 WMMA.
// ---------------------------------------------------------------------------
__global__ __launch_bounds__(128) void dense_wmma(const float* __restrict__ X,
                                                  const _Float16* __restrict__ Wh,
                                                  const float* __restrict__ b,
                                                  const float* __restrict__ a_ptr,
                                                  float* __restrict__ Y) {
  int wave = (blockIdx.x * blockDim.x + threadIdx.x) >> 5;
  int row0 = wave << 4;
  if (row0 >= NN) return;                     // wave-uniform: EXEC stays all-ones
  float a = a_ptr[0];
  v8f z = {};
  v8f acc[8];
#pragma unroll
  for (int jt = 0; jt < 8; ++jt) acc[jt] = z;
  for (int kc = 0; kc < DD; kc += 32) {
    v16h A = load_a_f32(X, row0, kc, 1.0f);
#pragma unroll
    for (int jt = 0; jt < 8; ++jt) {
      v16h B = load_b_h(Wh, jt << 4, kc);     // out = X@W^T -> B col j = W row j
      acc[jt] = __builtin_amdgcn_wmma_f32_16x16x32_f16(false, A, false, B,
                                                       (short)0, acc[jt], false, false);
    }
  }
  int l = threadIdx.x & 31;
  int Mo = (l >> 4) << 3, jl = l & 15;
#pragma unroll
  for (int jt = 0; jt < 8; ++jt) {
    int j = (jt << 4) + jl;
    float bj = b[j];
#pragma unroll
    for (int r = 0; r < 8; ++r) {
      float v = acc[jt][r] + bj;
      v = (v >= 0.0f) ? v : a * v;
      Y[(size_t)(row0 + Mo + r) * DD + j] = v;
    }
  }
}

// ---------------------------------------------------------------------------
// GCN feature: feat = (H * deg_out^-1/2) @ Wgcn^T  (row scale folded into A)
// ---------------------------------------------------------------------------
__global__ __launch_bounds__(128) void feat_wmma(const float* __restrict__ H,
                                                 const _Float16* __restrict__ Wgh,
                                                 const float* __restrict__ deg_out,
                                                 float* __restrict__ feat) {
  int wave = (blockIdx.x * blockDim.x + threadIdx.x) >> 5;
  int row0 = wave << 4;
  if (row0 >= NN) return;
  int l = threadIdx.x & 31;
  float scale = rsqrtf(fmaxf(deg_out[row0 + (l & 15)], 1.0f));
  v8f z = {};
  v8f acc[8];
#pragma unroll
  for (int jt = 0; jt < 8; ++jt) acc[jt] = z;
  for (int kc = 0; kc < DD; kc += 32) {
    v16h A = load_a_f32(H, row0, kc, scale);
#pragma unroll
    for (int jt = 0; jt < 8; ++jt) {
      v16h B = load_b_h(Wgh, jt << 4, kc);
      acc[jt] = __builtin_amdgcn_wmma_f32_16x16x32_f16(false, A, false, B,
                                                       (short)0, acc[jt], false, false);
    }
  }
  int Mo = (l >> 4) << 3, jl = l & 15;
#pragma unroll
  for (int jt = 0; jt < 8; ++jt)
#pragma unroll
    for (int r = 0; r < 8; ++r)
      feat[(size_t)(row0 + Mo + r) * DD + (jt << 4) + jl] = acc[jt][r];
}

// ---------------------------------------------------------------------------
// Edge scatter: agg[dst] += feat[src].  32 lanes/edge, float4 per lane.
// ---------------------------------------------------------------------------
__global__ void scatter_kernel(const float* __restrict__ feat, const int* __restrict__ src,
                               const int* __restrict__ dst, float* __restrict__ agg) {
  int t = blockIdx.x * blockDim.x + threadIdx.x;
  int e = t >> 5;
  if (e >= EE) return;
  int f4 = (t & 31) << 2;
  int s = src[e], d = dst[e];
  float4 v = *(const float4*)(feat + (size_t)s * DD + f4);
  float* ap = agg + (size_t)d * DD + f4;
  atomicAdd(ap + 0, v.x);
  atomicAdd(ap + 1, v.y);
  atomicAdd(ap + 2, v.z);
  atomicAdd(ap + 3, v.w);
}

// ---------------------------------------------------------------------------
// embed = agg * deg_in^-1/2 + b_gcn  (written straight into d_out region)
// ---------------------------------------------------------------------------
__global__ void embed_kernel(const float* __restrict__ agg, const float* __restrict__ deg_in,
                             const float* __restrict__ b_gcn, float* __restrict__ embed) {
  int t = blockIdx.x * blockDim.x + threadIdx.x;
  if (t >= NN * 32) return;
  int n = t >> 5, f4 = (t & 31) << 2;
  float rs = rsqrtf(fmaxf(deg_in[n], 1.0f));
  float4 v = *(const float4*)(agg + (size_t)n * DD + f4);
  float4 bg = *(const float4*)(b_gcn + f4);
  float4 o;
  o.x = v.x * rs + bg.x; o.y = v.y * rs + bg.y;
  o.z = v.z * rs + bg.z; o.w = v.w * rs + bg.w;
  *(float4*)(embed + (size_t)n * DD + f4) = o;
}

// ---------------------------------------------------------------------------
// Bilinear score: t = H @ Wbil (WMMA, kept in registers), score = <t, sigmoid(embed)>
// + b_bil + bias.  Cross-lane reduce matches C-layout (xor 1,2,4,8 within halves).
// ---------------------------------------------------------------------------
__global__ __launch_bounds__(128) void bilinear_wmma(const float* __restrict__ H,
                                                     const _Float16* __restrict__ Wbth,
                                                     const float* __restrict__ embed,
                                                     const float* __restrict__ bias,
                                                     const float* __restrict__ b_bil,
                                                     float* __restrict__ logit_out,
                                                     float* __restrict__ keys,
                                                     int* __restrict__ vals,
                                                     int write_keys) {
  int wave = (blockIdx.x * blockDim.x + threadIdx.x) >> 5;
  int row0 = wave << 4;
  if (row0 >= NN) return;
  v8f z = {};
  v8f acc[8];
#pragma unroll
  for (int jt = 0; jt < 8; ++jt) acc[jt] = z;
  for (int kc = 0; kc < DD; kc += 32) {
    v16h A = load_a_f32(H, row0, kc, 1.0f);
#pragma unroll
    for (int jt = 0; jt < 8; ++jt) {
      // t = H @ Wbil -> B col j = Wbil[:,j] = row j of pre-transposed Wbth
      v16h B = load_b_h(Wbth, jt << 4, kc);
      acc[jt] = __builtin_amdgcn_wmma_f32_16x16x32_f16(false, A, false, B,
                                                       (short)0, acc[jt], false, false);
    }
  }
  int l = threadIdx.x & 31;
  int half = l >> 4, jl = l & 15;
  float partial[8];
#pragma unroll
  for (int r = 0; r < 8; ++r) {
    int row = row0 + (half << 3) + r;
    float p = 0.0f;
#pragma unroll
    for (int jt = 0; jt < 8; ++jt) {
      float hc = sigm(embed[(size_t)row * DD + (jt << 4) + jl]);
      p += acc[jt][r] * hc;
    }
#pragma unroll
    for (int m = 1; m < 16; m <<= 1) p += __shfl_xor(p, m, 32);
    partial[r] = p;
  }
  if (jl == 0) {
    float bb = b_bil[0];
#pragma unroll
    for (int r = 0; r < 8; ++r) {
      int row = row0 + (half << 3) + r;
      float sc = partial[r] + bb + bias[row];
      logit_out[row] = sc;
      if (write_keys) { keys[row] = sigm(sc); vals[row] = row; }
    }
  }
}

// ---------------------------------------------------------------------------
// Bitonic sort pass (descending; idx-ascending tiebreak ~ stable argsort(-x))
// ---------------------------------------------------------------------------
__global__ void bitonic_step(float* __restrict__ key, int* __restrict__ val, int j, int kk) {
  int i = blockIdx.x * blockDim.x + threadIdx.x;
  int ixj = i ^ j;
  if (ixj > i && i < SORT_LEN) {
    float ka = key[i], kb = key[ixj];
    int va = val[i], vb = val[ixj];
    bool a_before_b = (ka > kb) || (ka == kb && va < vb);
    bool desc = ((i & kk) == 0);
    bool swap_ = desc ? (!a_before_b) : a_before_b;
    if (swap_) { key[i] = kb; key[ixj] = ka; val[i] = vb; val[ixj] = va; }
  }
}

// ---------------------------------------------------------------------------
// Emit top-k scores/indices and non-selected indices (ints stored bit-exact)
// ---------------------------------------------------------------------------
__global__ void select_kernel(const float* __restrict__ keys, const int* __restrict__ vals,
                              float* __restrict__ out) {
  int i = blockIdx.x * blockDim.x + threadIdx.x;
  if (i < KSEL) {
    out[2 * NN + i] = keys[i];                         // select_scores
    ((int*)out)[2 * NN + KSEL + i] = vals[i];          // select_idx
    ((int*)out)[2 * NN + 2 * KSEL + i] = vals[KSEL + i]; // non_select_idx
  }
}

// ---------------------------------------------------------------------------
extern "C" void kernel_launch(void* const* d_in, const int* in_sizes, int n_in,
                              void* d_out, int out_size, void* d_ws, size_t ws_size,
                              hipStream_t stream) {
  const float* h_pos    = (const float*)d_in[0];
  const float* h_neg    = (const float*)d_in[1];
  const float* bias_pos = (const float*)d_in[2];
  const float* bias_neg = (const float*)d_in[3];
  const float* W_dense  = (const float*)d_in[4];
  const float* b_dense  = (const float*)d_in[5];
  const float* a_prelu  = (const float*)d_in[6];
  const float* W_gcn    = (const float*)d_in[7];
  const float* b_gcn    = (const float*)d_in[8];
  const float* W_bil    = (const float*)d_in[9];
  const float* b_bil    = (const float*)d_in[10];
  const int*   src      = (const int*)d_in[11];
  const int*   dst      = (const int*)d_in[12];
  float* out = (float*)d_out;

  // workspace layout
  char* ws = (char*)d_ws;
  size_t off = 0;
  float* h_pos_act = (float*)(ws + off); off += (size_t)NN * DD * 4;
  float* h_neg_act = (float*)(ws + off); off += (size_t)NN * DD * 4;
  float* feat      = (float*)(ws + off); off += (size_t)NN * DD * 4;
  float* agg       = (float*)(ws + off); off += (size_t)NN * DD * 4;
  float* deg_out_b = (float*)(ws + off); off += (size_t)NN * 4;
  float* deg_in_b  = (float*)(ws + off); off += (size_t)NN * 4;
  float* keys      = (float*)(ws + off); off += (size_t)SORT_LEN * 4;
  int*   vals      = (int*)(ws + off);   off += (size_t)SORT_LEN * 4;
  _Float16* Wdh    = (_Float16*)(ws + off); off += (size_t)DD * DD * 2;
  _Float16* Wgh    = (_Float16*)(ws + off); off += (size_t)DD * DD * 2;
  _Float16* Wbth   = (_Float16*)(ws + off); off += (size_t)DD * DD * 2;

  float* logit_pos = out;               // [0, N)
  float* logit_neg = out + NN;          // [N, 2N)
  float* embed_out = out + 2 * NN + 3 * KSEL; // embed region

  // 1) prep
  prep_kernel<<<(NN * DD + 255) / 256, 256, 0, stream>>>(
      W_dense, W_gcn, W_bil, Wdh, Wgh, Wbth, agg, deg_out_b, deg_in_b, keys, vals);
  // 2) degrees
  degree_kernel<<<(EE + 255) / 256, 256, 0, stream>>>(src, dst, deg_out_b, deg_in_b);
  // 3) dense layers (3125 waves of 16 rows each; 128-thread blocks = 4 waves)
  const int gemm_blocks = (NN / 16 * 32 + 127) / 128;
  dense_wmma<<<gemm_blocks, 128, 0, stream>>>(h_pos, Wdh, b_dense, a_prelu, h_pos_act);
  dense_wmma<<<gemm_blocks, 128, 0, stream>>>(h_neg, Wdh, b_dense, a_prelu, h_neg_act);
  // 4) GCN feature GEMM
  feat_wmma<<<gemm_blocks, 128, 0, stream>>>(h_pos_act, Wgh, deg_out_b, feat);
  // 5) edge scatter (bandwidth-dominant)
  scatter_kernel<<<(EE * 32 + 255) / 256, 256, 0, stream>>>(feat, src, dst, agg);
  // 6) embed finalize (straight into d_out)
  embed_kernel<<<(NN * 32 + 255) / 256, 256, 0, stream>>>(agg, deg_in_b, b_gcn, embed_out);
  // 7) bilinear scores (pos also seeds sort keys)
  bilinear_wmma<<<gemm_blocks, 128, 0, stream>>>(h_pos_act, Wbth, embed_out, bias_pos,
                                                 b_bil, logit_pos, keys, vals, 1);
  bilinear_wmma<<<gemm_blocks, 128, 0, stream>>>(h_neg_act, Wbth, embed_out, bias_neg,
                                                 b_bil, logit_neg, keys, vals, 0);
  // 8) bitonic argsort, descending
  for (int kk = 2; kk <= SORT_LEN; kk <<= 1)
    for (int j = kk >> 1; j > 0; j >>= 1)
      bitonic_step<<<SORT_LEN / 256, 256, 0, stream>>>(keys, vals, j, kk);
  // 9) emit top-k split
  select_kernel<<<(KSEL + 255) / 256, 256, 0, stream>>>(keys, vals, out);
}
